// GraphNeuralSDE_55740085567595
// MI455X (gfx1250) — compile-verified, matrix operand order checked by reading, not run on hardware
//
#include <hip/hip_runtime.h>
#include <cstddef>

// Problem constants (match reference)
#define NN 8192          // nodes
#define DD 16            // state dim
#define HH 64            // hidden dim
#define BB 4             // batch
#define NCOL 64          // BB*DD = GEMM column count
#define KSPLIT 4
#define KCHUNK (NN / KSPLIT)          // 2048
#define KB_PER_CHUNK (KCHUNK / 32)    // 64 k-blocks
#define ROWS_PER_WAVE 32
#define WAVES_PER_WG 4
#define ROWS_PER_WG (ROWS_PER_WAVE * WAVES_PER_WG)  // 128

typedef __attribute__((ext_vector_type(16))) __bf16        v16bf;
typedef __attribute__((ext_vector_type(8)))  float         v8f;
typedef __attribute__((ext_vector_type(4)))  float         v4f;
typedef __attribute__((ext_vector_type(4)))  unsigned int  v4u;

union BF16Frag {            // 8 VGPRs worth of bf16 operand data
  unsigned int u[8];
  v16bf        v;
};

struct AStage {             // 16 fp32 of one A fragment (per lane)
  v4f a0, a1, a2, a3;
};

__device__ __forceinline__ unsigned short f2bf(float f) {
  union { float f; unsigned int u; } c; c.f = f;
  unsigned int u = c.u;
  u += 0x7FFFu + ((u >> 16) & 1u);      // round-to-nearest-even
  return (unsigned short)(u >> 16);
}
__device__ __forceinline__ unsigned int pk2(unsigned short lo, unsigned short hi) {
  return (unsigned int)lo | ((unsigned int)hi << 16);
}
__device__ __forceinline__ void loadA(AStage& s, const float* p) {
  s.a0 = *(const v4f*)(p + 0);          // global_load_b128 x4
  s.a1 = *(const v4f*)(p + 4);
  s.a2 = *(const v4f*)(p + 16);
  s.a3 = *(const v4f*)(p + 20);
}
__device__ __forceinline__ void cvtA(BF16Frag& f, const AStage& s) {
  f.u[0] = pk2(f2bf(s.a0.x), f2bf(s.a0.y));
  f.u[1] = pk2(f2bf(s.a0.z), f2bf(s.a0.w));
  f.u[2] = pk2(f2bf(s.a1.x), f2bf(s.a1.y));
  f.u[3] = pk2(f2bf(s.a1.z), f2bf(s.a1.w));
  f.u[4] = pk2(f2bf(s.a2.x), f2bf(s.a2.y));
  f.u[5] = pk2(f2bf(s.a2.z), f2bf(s.a2.w));
  f.u[6] = pk2(f2bf(s.a3.x), f2bf(s.a3.y));
  f.u[7] = pk2(f2bf(s.a3.z), f2bf(s.a3.w));
}
__device__ __forceinline__ void loadFrag(BF16Frag& d, const unsigned short* p) {
  v4u b0 = *(const v4u*)p;          // global_load_b128
  v4u b1 = *(const v4u*)(p + 8);    // global_load_b128
  d.u[0] = b0.x; d.u[1] = b0.y; d.u[2] = b0.z; d.u[3] = b0.w;
  d.u[4] = b1.x; d.u[5] = b1.y; d.u[6] = b1.z; d.u[7] = b1.w;
}

// Stage one k-block (2 A fragments fp32 + 4 B fragments bf16) into registers
__device__ __forceinline__ void stageK(
    AStage& A0, AStage& A1,
    BF16Frag& B0, BF16Frag& B1, BF16Frag& B2, BF16Frag& B3,
    const float* aPtr0, const float* aPtr1, const unsigned short* bPtr) {
  loadA(A0, aPtr0);
  loadA(A1, aPtr1);
  loadFrag(B0, bPtr + 0 * 512);
  loadFrag(B1, bPtr + 1 * 512);
  loadFrag(B2, bPtr + 2 * 512);
  loadFrag(B3, bPtr + 3 * 512);
}

// 2 A fragments x 4 B fragments -> 8 WMMAs
__device__ __forceinline__ void doWmma8(
    const AStage& A0, const AStage& A1,
    v16bf b0, v16bf b1, v16bf b2, v16bf b3,
    v8f& c00, v8f& c01, v8f& c02, v8f& c03,
    v8f& c10, v8f& c11, v8f& c12, v8f& c13) {
  BF16Frag af0, af1;
  cvtA(af0, A0);
  cvtA(af1, A1);
  c00 = __builtin_amdgcn_wmma_f32_16x16x32_bf16(false, af0.v, false, b0, (short)0, c00, false, false);
  c01 = __builtin_amdgcn_wmma_f32_16x16x32_bf16(false, af0.v, false, b1, (short)0, c01, false, false);
  c02 = __builtin_amdgcn_wmma_f32_16x16x32_bf16(false, af0.v, false, b2, (short)0, c02, false, false);
  c03 = __builtin_amdgcn_wmma_f32_16x16x32_bf16(false, af0.v, false, b3, (short)0, c03, false, false);
  c10 = __builtin_amdgcn_wmma_f32_16x16x32_bf16(false, af1.v, false, b0, (short)0, c10, false, false);
  c11 = __builtin_amdgcn_wmma_f32_16x16x32_bf16(false, af1.v, false, b1, (short)0, c11, false, false);
  c12 = __builtin_amdgcn_wmma_f32_16x16x32_bf16(false, af1.v, false, b2, (short)0, c12, false, false);
  c13 = __builtin_amdgcn_wmma_f32_16x16x32_bf16(false, af1.v, false, b3, (short)0, c13, false, false);
}

// ---------------------------------------------------------------------------
// Kernel 1: pack Hcat (N x 64, col = b*16+d) into bf16 WMMA B-fragments.
//   lane L (L<16): col = cb*16+L, K = kb*32 + {0..7} then {16..23}
//   lane L (L>=16): col = cb*16+(L-16), K = kb*32 + {8..15} then {24..31}
// ---------------------------------------------------------------------------
__global__ __launch_bounds__(256) void gnsde_pack_b(
    const float* __restrict__ h, unsigned short* __restrict__ Bfrag) {
  const int tid  = blockIdx.x * 256 + threadIdx.x;   // 32768 total
  const int lane = tid & 31;
  const int frag = tid >> 5;                         // 0..1023
  const int kb   = frag >> 2;                        // k-block of 32
  const int cb   = frag & 3;                         // col block == batch index
  const int d    = lane & 15;
  const int base = (lane >> 4) * 8;
  const float* hb = h + (size_t)cb * (NN * DD);
  unsigned short* op = Bfrag + (size_t)frag * 512 + lane * 16;
#pragma unroll
  for (int j = 0; j < 8; ++j) {
    const int k1 = kb * 32 + base + j;
    const int k2 = k1 + 16;
    op[j]     = f2bf(hb[(size_t)k1 * DD + d]);
    op[8 + j] = f2bf(hb[(size_t)k2 * DD + d]);
  }
}

// ---------------------------------------------------------------------------
// Kernel 2: Out_part[ky] = A[:, kchunk] @ Hcat[kchunk, :]  via bf16 WMMA.
// Wave = 32 rows x 64 cols (8 accumulators, 2 A frags sharing 4 B frags).
// Ping-pong register double-buffering (unroll x2, two named buffer sets):
// no rotation movs, loads for block k+1 overlap WMMAs of block k, and the
// off-buffer conversion VALU fills WMMA->VALU hazard slots. The fp32 A
// stream (256 MiB @ 23.3 TB/s ~= 11.5 us) is the intended bottleneck.
// ---------------------------------------------------------------------------
__global__ __launch_bounds__(WAVES_PER_WG * 32) void gnsde_gemm(
    const float* __restrict__ A,
    const unsigned short* __restrict__ Bfrag,
    float* __restrict__ hg_part) {
  const int wave    = threadIdx.x >> 5;
  const int lane    = threadIdx.x & 31;
  const int rowBase = blockIdx.x * ROWS_PER_WG + wave * ROWS_PER_WAVE;
  const int ky      = blockIdx.y;
  const int kBegin  = ky * KCHUNK;
  const int hi      = lane >> 4;       // 0 or 1
  const int lo      = lane & 15;
  const int koff    = hi * 8;

  v8f c00 = {}, c01 = {}, c02 = {}, c03 = {};
  v8f c10 = {}, c11 = {}, c12 = {}, c13 = {};

  const float* aPtr0 = A + (size_t)(rowBase + lo) * NN + kBegin + koff;
  const float* aPtr1 = aPtr0 + (size_t)16 * NN;
  const unsigned short* bPtr =
      Bfrag + ((size_t)(kBegin >> 5) * 4) * 512 + lane * 16;

  // Two full register buffer sets (ping-pong; no rotation copies)
  AStage   A0a, A1a, A0b, A1b;
  BF16Frag B0a, B1a, B2a, B3a, B0b, B1b, B2b, B3b;

  // Prologue: stage k-block 0 into set 'a'
  stageK(A0a, A1a, B0a, B1a, B2a, B3a, aPtr0, aPtr1, bPtr);

#pragma unroll 1
  for (int kb = 0; kb < KB_PER_CHUNK / 2 - 1; ++kb) {
    // stage odd block into 'b', compute even block from 'a'
    aPtr0 += 32; aPtr1 += 32; bPtr += 4 * 512;
    stageK(A0b, A1b, B0b, B1b, B2b, B3b, aPtr0, aPtr1, bPtr);
    __builtin_prefetch(aPtr0 + 256, 0, 1);   // warm GL2 ~8 k-blocks ahead
    __builtin_prefetch(aPtr1 + 256, 0, 1);
    doWmma8(A0a, A1a, B0a.v, B1a.v, B2a.v, B3a.v,
            c00, c01, c02, c03, c10, c11, c12, c13);

    // stage next even block into 'a', compute odd block from 'b'
    aPtr0 += 32; aPtr1 += 32; bPtr += 4 * 512;
    stageK(A0a, A1a, B0a, B1a, B2a, B3a, aPtr0, aPtr1, bPtr);
    __builtin_prefetch(aPtr0 + 256, 0, 1);
    __builtin_prefetch(aPtr1 + 256, 0, 1);
    doWmma8(A0b, A1b, B0b.v, B1b.v, B2b.v, B3b.v,
            c00, c01, c02, c03, c10, c11, c12, c13);
  }
  // Epilogue: blocks 62 (in 'a') and 63
  aPtr0 += 32; aPtr1 += 32; bPtr += 4 * 512;
  stageK(A0b, A1b, B0b, B1b, B2b, B3b, aPtr0, aPtr1, bPtr);
  doWmma8(A0a, A1a, B0a.v, B1a.v, B2a.v, B3a.v,
          c00, c01, c02, c03, c10, c11, c12, c13);
  doWmma8(A0b, A1b, B0b.v, B1b.v, B2b.v, B3b.v,
          c00, c01, c02, c03, c10, c11, c12, c13);

  // C/D layout: VGPR r -> M = r (lanes 0-15) or r+8 (lanes 16-31), N = lane%16
  const int rshift = hi * 8;
  float* op = hg_part + ((size_t)ky * NN + rowBase) * NCOL;
#pragma unroll
  for (int r = 0; r < 8; ++r) {
    const size_t rr = (size_t)(r + rshift) * NCOL + lo;
    op[rr + 0]  = c00[r];
    op[rr + 16] = c01[r];
    op[rr + 32] = c02[r];
    op[rr + 48] = c03[r];
  }
#pragma unroll
  for (int r = 0; r < 8; ++r) {
    const size_t rr = (size_t)(16 + r + rshift) * NCOL + lo;
    op[rr + 0]  = c10[r];
    op[rr + 16] = c11[r];
    op[rr + 32] = c12[r];
    op[rr + 48] = c13[r];
  }
}

// ---------------------------------------------------------------------------
// Kernel 3: reduce K-partials + fused drift/diffusion MLPs.
// One thread per (b, n). Weights staged in LDS; all accumulators in regs.
// ---------------------------------------------------------------------------
__global__ __launch_bounds__(256) void gnsde_mlp(
    const float* __restrict__ h,   const float* __restrict__ t,
    const float* __restrict__ W1f, const float* __restrict__ b1f,
    const float* __restrict__ W2f, const float* __restrict__ b2f,
    const float* __restrict__ W1g, const float* __restrict__ b1g,
    const float* __restrict__ W2g, const float* __restrict__ b2g,
    const float* __restrict__ hg_part, float* __restrict__ out) {
  __shared__ float sW1f[17 * 64], sW1g[17 * 64];
  __shared__ float sW2f[64 * 16], sW2g[64 * 16];
  __shared__ float sb1f[64], sb1g[64], sb2f[16], sb2g[16];

  for (int i = threadIdx.x; i < 17 * 64; i += 256) { sW1f[i] = W1f[i]; sW1g[i] = W1g[i]; }
  for (int i = threadIdx.x; i < 64 * 16; i += 256) { sW2f[i] = W2f[i]; sW2g[i] = W2g[i]; }
  if (threadIdx.x < 64) { sb1f[threadIdx.x] = b1f[threadIdx.x]; sb1g[threadIdx.x] = b1g[threadIdx.x]; }
  if (threadIdx.x < 16) { sb2f[threadIdx.x] = b2f[threadIdx.x]; sb2g[threadIdx.x] = b2g[threadIdx.x]; }
  __syncthreads();

  const int gid = blockIdx.x * 256 + threadIdx.x;  // 0..32767
  const int b = gid >> 13;
  const int n = gid & (NN - 1);
  const float tv = t[0];

  float hgv[16], hrv[16];
#pragma unroll
  for (int d = 0; d < 16; ++d) {
    float s = 0.f;
#pragma unroll
    for (int ky = 0; ky < KSPLIT; ++ky)
      s += hg_part[((size_t)ky * NN + n) * NCOL + b * 16 + d];
    hgv[d] = s;
    hrv[d] = h[(size_t)b * (NN * DD) + (size_t)n * DD + d];
  }

  float sf[16], sg[16];
#pragma unroll
  for (int d = 0; d < 16; ++d) { sf[d] = sb2f[d]; sg[d] = sb2g[d]; }

  for (int j = 0; j < 64; ++j) {
    float af = sb1f[j] + tv * sW1f[16 * 64 + j];
    float ag = sb1g[j] + tv * sW1g[16 * 64 + j];
#pragma unroll
    for (int d = 0; d < 16; ++d) {
      af = fmaf(hgv[d], sW1f[d * 64 + j], af);
      ag = fmaf(hrv[d], sW1g[d * 64 + j], ag);
    }
    const float z1 = tanhf(af);
    const float z2 = 1.f / (1.f + __expf(-ag));
#pragma unroll
    for (int d = 0; d < 16; ++d) {
      sf[d] = fmaf(z1, sW2f[j * 16 + d], sf[d]);
      sg[d] = fmaf(z2, sW2g[j * 16 + d], sg[d]);
    }
  }

#pragma unroll
  for (int d = 0; d < 16; ++d) {
    const size_t oo = (size_t)b * (NN * DD) + (size_t)n * DD + d;
    out[oo] = sf[d];                                           // drift
    out[(size_t)BB * NN * DD + oo] = 0.1f / (1.f + __expf(-sg[d]));  // diffusion
  }
}

// ---------------------------------------------------------------------------
extern "C" void kernel_launch(void* const* d_in, const int* in_sizes, int n_in,
                              void* d_out, int out_size, void* d_ws, size_t ws_size,
                              hipStream_t stream) {
  const float* h   = (const float*)d_in[0];
  const float* t   = (const float*)d_in[1];
  const float* A   = (const float*)d_in[2];
  const float* W1f = (const float*)d_in[3];
  const float* b1f = (const float*)d_in[4];
  const float* W2f = (const float*)d_in[5];
  const float* b2f = (const float*)d_in[6];
  const float* W1g = (const float*)d_in[7];
  const float* b1g = (const float*)d_in[8];
  const float* W2g = (const float*)d_in[9];
  const float* b2g = (const float*)d_in[10];
  float* out = (float*)d_out;

  // workspace: [0, 1 MiB) bf16 B-fragments; [1 MiB, 9 MiB) fp32 K-partials
  unsigned short* Bfrag = (unsigned short*)d_ws;
  float* hg_part = (float*)((char*)d_ws + (size_t)NN * NCOL * sizeof(unsigned short));

  gnsde_pack_b<<<dim3(128), dim3(256), 0, stream>>>(h, Bfrag);
  gnsde_gemm<<<dim3(NN / ROWS_PER_WG, KSPLIT), dim3(WAVES_PER_WG * 32), 0, stream>>>(
      A, Bfrag, hg_part);
  gnsde_mlp<<<dim3((BB * NN) / 256), dim3(256), 0, stream>>>(
      h, t, W1f, b1f, W2f, b2f, W1g, b1g, W2g, b2g, hg_part, out);
}
